// GraphNetwork_50122268345054
// MI455X (gfx1250) — matrix operand order repaired and checked by reading, not compiled, not run on hardware
//
#include <hip/hip_runtime.h>
#include <hip/hip_bf16.h>
#include <math.h>

// ---------------------------------------------------------------------------
// Types for CDNA5 WMMA (wave32, V_WMMA_F32_16X16X32_F16)
// ---------------------------------------------------------------------------
typedef __attribute__((ext_vector_type(16))) _Float16 v16h;
typedef __attribute__((ext_vector_type(8)))  _Float16 v8h;
typedef __attribute__((ext_vector_type(8)))  float    v8f;

#define C_S 0.3826834323650898f   // sin(pi/8)
#define C_X 0.9238795325112867f   // cos(pi/8)

// A-fragment K index for 16-bit A 16x32 (ISA 7.12.2)
__device__ __forceinline__ int klocalA(int j, int half) {
    return ((j < 8) ? j : j + 8) + 8 * half;
}

__device__ __forceinline__ float silu_f(float x) { return x / (1.f + __expf(-x)); }
__device__ __forceinline__ float sigm_f(float x) { return 1.f / (1.f + __expf(-x)); }

// ---------------------------------------------------------------------------
// Weight packer: f32 [K, srcStride] (uses V cols) -> f16 WMMA-B fragments.
// B layout: lane=(n=lane&15, half=lane>>4), elem j = B(k = (j<8?j:j+8)+8*half+32*kIter,
// n+16*vt).  Fragment order ((kIter*numVT+vt)*32+lane)*16+j.
// ---------------------------------------------------------------------------
__global__ void pack_weight(const float* __restrict__ src, _Float16* __restrict__ dst,
                            int K, int V, int srcStride, int Kpad) {
    int idx = blockIdx.x * blockDim.x + threadIdx.x;
    int total = (Kpad >> 5) * (V >> 4) * 32;
    if (idx >= total) return;
    int lane = idx & 31;
    int t = idx >> 5;
    int numVT = V >> 4;
    int vt = t % numVT;
    int kIter = t / numVT;
    int half = lane >> 4;
    int col = (lane & 15) + (vt << 4);
#pragma unroll
    for (int j = 0; j < 16; ++j) {
        int k = (kIter << 5) + ((j < 8) ? j : j + 8) + 8 * half;
        int kc = (k < K) ? k : (K - 1);          // clamp -> unconditional load
        float v = src[(long)kc * srcStride + col];
        v = (k < K) ? v : 0.f;                   // branch-free zero pad
        dst[(long)idx * 16 + j] = (_Float16)v;
    }
}

// ---------------------------------------------------------------------------
// Elementwise helpers
// ---------------------------------------------------------------------------
__global__ void zero_kernel(float* p, int n) {
    int i = blockIdx.x * blockDim.x + threadIdx.x;
    if (i < n) p[i] = 0.f;
}
__global__ void inv_sqrt_kernel(const float* __restrict__ deg, float* __restrict__ inv, int n) {
    int i = blockIdx.x * blockDim.x + threadIdx.x;
    if (i < n) inv[i] = rsqrtf(deg[i]);
}
template <int C>
__global__ void scale_rows_kernel(float* __restrict__ p, int n, const float* __restrict__ s) {
    int i = blockIdx.x * blockDim.x + threadIdx.x;
    if (i < n) p[i] *= s[i / C];
}
__global__ void gate_v_kernel(float* __restrict__ v1, const float* __restrict__ g1, int n) {
    int i = blockIdx.x * blockDim.x + threadIdx.x;
    if (i < n) {
        float g = g1[i];
        v1[3 * i + 0] *= g;
        v1[3 * i + 1] *= g;
        v1[3 * i + 2] *= g;
    }
}

// ---------------------------------------------------------------------------
// zz fragment load: two contiguous 8-float runs -> 4x float4, convert to f16
// ---------------------------------------------------------------------------
__device__ __forceinline__ void load_zfrag(const float* __restrict__ zz, int node, int half,
                                           _Float16* zfh) {
    const float4* zr1 = reinterpret_cast<const float4*>(zz + (long)node * 32 + 8 * half);
    const float4* zr2 = reinterpret_cast<const float4*>(zz + (long)node * 32 + 16 + 8 * half);
    float4 z0 = zr1[0], z1 = zr1[1], z2 = zr2[0], z3 = zr2[1];
    zfh[0]=(_Float16)z0.x; zfh[1]=(_Float16)z0.y; zfh[2]=(_Float16)z0.z; zfh[3]=(_Float16)z0.w;
    zfh[4]=(_Float16)z1.x; zfh[5]=(_Float16)z1.y; zfh[6]=(_Float16)z1.z; zfh[7]=(_Float16)z1.w;
    zfh[8]=(_Float16)z2.x; zfh[9]=(_Float16)z2.y; zfh[10]=(_Float16)z2.z; zfh[11]=(_Float16)z2.w;
    zfh[12]=(_Float16)z3.x; zfh[13]=(_Float16)z3.y; zfh[14]=(_Float16)z3.z; zfh[15]=(_Float16)z3.w;
}

__device__ __forceinline__ float fctp_post(float val, int mode, float sc, float mk) {
    if (mode == 1) return val * sc;
    if (mode >= 2) {
        float comb = C_S * mk + C_X * val;
        if (mode == 2) return silu_f(comb);
        if (mode == 3) return sigm_f(comb);
        return comb;
    }
    return val;
}

// ---------------------------------------------------------------------------
// Embedding GEMM:  out = relu( relu(X[N,118]) @ W[118,32] + b )  (Kpad=128)
// ---------------------------------------------------------------------------
__global__ void embed_kernel(const float* __restrict__ X, const float* __restrict__ bias,
                             const _Float16* __restrict__ Wfrag, float* __restrict__ out) {
    int tid = threadIdx.x;
    int m = tid & 15, half = tid >> 4;
    int tile = blockIdx.x;
    int node = tile * 16 + m;
    const float* xrow = X + (long)node * 118;
    const v16h* W = reinterpret_cast<const v16h*>(Wfrag);
    v8f acc[2];
#pragma unroll
    for (int vt = 0; vt < 2; ++vt) { v8f z = {}; acc[vt] = z; }
    union { v16h v; _Float16 h[16]; } a;
#pragma unroll
    for (int kI = 0; kI < 4; ++kI) {
#pragma unroll
        for (int j = 0; j < 16; ++j) {
            int k = kI * 32 + klocalA(j, half);
            int kc = (k < 118) ? k : 117;
            float xv = xrow[kc];
            xv = (k < 118) ? xv : 0.f;
            a.h[j] = (_Float16)(xv > 0.f ? xv : 0.f);
        }
#pragma unroll
        for (int vt = 0; vt < 2; ++vt)
            acc[vt] = __builtin_amdgcn_wmma_f32_16x16x32_f16(
                false, a.v, false, W[(kI * 2 + vt) * 32 + tid], (short)0, acc[vt], false, false);
    }
#pragma unroll
    for (int vt = 0; vt < 2; ++vt) {
        union { v8f v; float f[8]; } c; c.v = acc[vt];
#pragma unroll
        for (int r = 0; r < 8; ++r) {
            int row = r + 8 * half;
            int col = vt * 16 + m;
            float val = c.f[r] + bias[col];
            out[(long)(tile * 16 + row) * 32 + col] = val > 0.f ? val : 0.f;
        }
    }
}

// ---------------------------------------------------------------------------
// Fused fctp: ONE A-build (A = P⊗zz) feeds NOUT weight sets (V=32 each).
// Per output: rowStride rs_o, mode_o (0 raw, 1 *scale, 2 silu comb, 3 sigm comb)
// ---------------------------------------------------------------------------
template <int NOUT>
__global__ void fctp_multi_kernel(const float* __restrict__ P, const float* __restrict__ zz,
                                  const _Float16* __restrict__ Wf0,
                                  const _Float16* __restrict__ Wf1,
                                  const _Float16* __restrict__ Wf2,
                                  float* __restrict__ out0, float* __restrict__ out1,
                                  float* __restrict__ out2,
                                  int rs0, int rs1, int rs2,
                                  int mode0, int mode1, int mode2,
                                  const float* __restrict__ scaleVec,
                                  const float* __restrict__ mk0, const float* __restrict__ mk1,
                                  float rnorm) {
    int tid = threadIdx.x;
    int m = tid & 15, half = tid >> 4;
    int tile = blockIdx.x;
    int node = tile * 16 + m;

    _Float16 zfh[16];
    load_zfrag(zz, node, half, zfh);

    const v16h* W[3] = {reinterpret_cast<const v16h*>(Wf0),
                        reinterpret_cast<const v16h*>(Wf1),
                        reinterpret_cast<const v16h*>(Wf2)};
    float* outp[3] = {out0, out1, out2};
    int rs[3] = {rs0, rs1, rs2};
    int mode[3] = {mode0, mode1, mode2};
    const float* mk[3] = {mk0, mk1, nullptr};

    v8f acc[NOUT][2];
#pragma unroll
    for (int o = 0; o < NOUT; ++o)
#pragma unroll
        for (int vt = 0; vt < 2; ++vt) { v8f zv = {}; acc[o][vt] = zv; }

    const float* prow = P + (long)node * 32;
    for (int u = 0; u < 32; ++u) {
        _Float16 ph = (_Float16)prow[u];
        union { v16h v; _Float16 h[16]; } a;
#pragma unroll
        for (int j = 0; j < 16; ++j) a.h[j] = ph * zfh[j];
#pragma unroll
        for (int o = 0; o < NOUT; ++o)
#pragma unroll
            for (int vt = 0; vt < 2; ++vt)
                acc[o][vt] = __builtin_amdgcn_wmma_f32_16x16x32_f16(
                    false, a.v, false, W[o][(u * 2 + vt) * 32 + tid], (short)0,
                    acc[o][vt], false, false);
    }

#pragma unroll
    for (int o = 0; o < NOUT; ++o) {
#pragma unroll
        for (int vt = 0; vt < 2; ++vt) {
            union { v8f v; float f[8]; } c; c.v = acc[o][vt];
#pragma unroll
            for (int r = 0; r < 8; ++r) {
                int row = r + 8 * half;
                int nodeR = tile * 16 + row;
                int col = vt * 16 + m;
                float sc = (mode[o] == 1) ? scaleVec[nodeR] : 0.f;
                float mv = (mode[o] >= 2) ? mk[o][(long)nodeR * 32 + col] : 0.f;
                outp[o][(long)nodeR * rs[o] + col] = fctp_post(c.f[r] * rnorm, mode[o], sc, mv);
            }
        }
    }
}

// ---------------------------------------------------------------------------
// Vector fctp: 3 cartesian channels share zz fragment + W stream.
// P,out are [N,32,3]; mode 0 raw, 1 *scaleVec[n].
// ---------------------------------------------------------------------------
__global__ void fctp_vec3_kernel(const float* __restrict__ P, const float* __restrict__ zz,
                                 const _Float16* __restrict__ Wfrag, float* __restrict__ out,
                                 const float* __restrict__ scaleVec, float rnorm, int mode) {
    int tid = threadIdx.x;
    int m = tid & 15, half = tid >> 4;
    int tile = blockIdx.x;
    int node = tile * 16 + m;

    _Float16 zfh[16];
    load_zfrag(zz, node, half, zfh);

    const v16h* W = reinterpret_cast<const v16h*>(Wfrag);
    const float* prow = P + (long)node * 96;

    v8f acc[3][2];
#pragma unroll
    for (int c = 0; c < 3; ++c)
#pragma unroll
        for (int vt = 0; vt < 2; ++vt) { v8f zv = {}; acc[c][vt] = zv; }

    for (int u = 0; u < 32; ++u) {
        _Float16 ph[3];
#pragma unroll
        for (int c = 0; c < 3; ++c) ph[c] = (_Float16)prow[u * 3 + c];
        v16h b0 = W[(u * 2 + 0) * 32 + tid];
        v16h b1 = W[(u * 2 + 1) * 32 + tid];
#pragma unroll
        for (int c = 0; c < 3; ++c) {
            union { v16h v; _Float16 h[16]; } a;
#pragma unroll
            for (int j = 0; j < 16; ++j) a.h[j] = ph[c] * zfh[j];
            acc[c][0] = __builtin_amdgcn_wmma_f32_16x16x32_f16(
                false, a.v, false, b0, (short)0, acc[c][0], false, false);
            acc[c][1] = __builtin_amdgcn_wmma_f32_16x16x32_f16(
                false, a.v, false, b1, (short)0, acc[c][1], false, false);
        }
    }

#pragma unroll
    for (int c = 0; c < 3; ++c)
#pragma unroll
        for (int vt = 0; vt < 2; ++vt) {
            union { v8f v; float f[8]; } cc; cc.v = acc[c][vt];
#pragma unroll
            for (int r = 0; r < 8; ++r) {
                int row = r + 8 * half;
                int nodeR = tile * 16 + row;
                int col = vt * 16 + m;
                float val = cc.f[r] * rnorm;
                if (mode == 1) val *= scaleVec[nodeR];
                out[((long)nodeR * 32 + col) * 3 + c] = val;
            }
        }
}

// ---------------------------------------------------------------------------
// Single fctp (used for the final V=64 contraction with mask combine)
// ---------------------------------------------------------------------------
template <int NVT>
__global__ void fctp_kernel(const float* __restrict__ P, int pRowStride, int U,
                            const float* __restrict__ zz,
                            const _Float16* __restrict__ Wfrag,
                            float* __restrict__ out, int outRowStride,
                            const float* __restrict__ maskBuf, int maskRowStride,
                            float rnorm, int mode) {
    int tid = threadIdx.x;
    int m = tid & 15, half = tid >> 4;
    int tile = blockIdx.x;
    int node = tile * 16 + m;

    _Float16 zfh[16];
    load_zfrag(zz, node, half, zfh);

    v8f acc[NVT];
#pragma unroll
    for (int vt = 0; vt < NVT; ++vt) { v8f z = {}; acc[vt] = z; }

    const v16h* W = reinterpret_cast<const v16h*>(Wfrag);
    const float* prow = P + (long)node * pRowStride;

    for (int u = 0; u < U; ++u) {
        _Float16 ph = (_Float16)prow[u];
        union { v16h v; _Float16 h[16]; } a;
#pragma unroll
        for (int j = 0; j < 16; ++j) a.h[j] = ph * zfh[j];
#pragma unroll
        for (int vt = 0; vt < NVT; ++vt)
            acc[vt] = __builtin_amdgcn_wmma_f32_16x16x32_f16(
                false, a.v, false, W[(u * NVT + vt) * 32 + tid], (short)0, acc[vt], false, false);
    }

#pragma unroll
    for (int vt = 0; vt < NVT; ++vt) {
        union { v8f v; float f[8]; } c; c.v = acc[vt];
#pragma unroll
        for (int r = 0; r < 8; ++r) {
            int row = r + 8 * half;
            int nodeR = tile * 16 + row;
            int col = vt * 16 + m;
            float mv = (mode >= 2) ? maskBuf[(long)nodeR * maskRowStride + col] : 0.f;
            out[(long)nodeR * outRowStride + col] = fctp_post(c.f[r] * rnorm, mode, 0.f, mv);
        }
    }
}

// ---------------------------------------------------------------------------
// Radial MLP for one 16-edge tile (both edge stages):
// emb[16x10] --W0--> silu --> h[16x64] (f16 LDS) --W1--> w[16x64] (f32 LDS)
// ---------------------------------------------------------------------------
__device__ __forceinline__ void radial_and_y1(int tile, int tid,
                                              const float* __restrict__ elen,
                                              const float* __restrict__ evec,
                                              const _Float16* __restrict__ w0p,
                                              const _Float16* __restrict__ w1p,
                                              _Float16 (*hbuf)[64], float (*wbuf)[64],
                                              float (*y1s)[3]) {
    int m = tid & 15, half = tid >> 4;
    int e = tile * 16 + m;
    float len = elen[e];

    union { v16h v; _Float16 h[16]; } a;
#pragma unroll
    for (int j = 0; j < 16; ++j) {
        int k = klocalA(j, half);
        float val = 0.f;
        if (k < 10) {
            float d = len * 1.8f - (float)k;
            val = __expf(-d * d) * (1.0f / 1.12f);
        }
        a.h[j] = (_Float16)val;
    }

    const v16h* W0 = reinterpret_cast<const v16h*>(w0p);
    const v16h* W1 = reinterpret_cast<const v16h*>(w1p);

    v8f acc[4];
#pragma unroll
    for (int vt = 0; vt < 4; ++vt) {
        v8f z = {};
        acc[vt] = __builtin_amdgcn_wmma_f32_16x16x32_f16(
            false, a.v, false, W0[vt * 32 + tid], (short)0, z, false, false);
    }
#pragma unroll
    for (int vt = 0; vt < 4; ++vt) {
        union { v8f v; float f[8]; } c; c.v = acc[vt];
#pragma unroll
        for (int r = 0; r < 8; ++r)
            hbuf[r + 8 * half][vt * 16 + m] = (_Float16)silu_f(c.f[r]);
    }
    if (half == 0) {
        float invl = 1.7320508075688772f / (len + 1e-12f);
#pragma unroll
        for (int c2 = 0; c2 < 3; ++c2) y1s[m][c2] = evec[e * 3 + c2] * invl;
    }
    __syncthreads();

    v8f acc2[4];
#pragma unroll
    for (int vt = 0; vt < 4; ++vt) { v8f z = {}; acc2[vt] = z; }
#pragma unroll
    for (int kI = 0; kI < 2; ++kI) {
        v8h r1 = *reinterpret_cast<const v8h*>(&hbuf[m][kI * 32 + 8 * half]);
        v8h r2 = *reinterpret_cast<const v8h*>(&hbuf[m][kI * 32 + 16 + 8 * half]);
#pragma unroll
        for (int j = 0; j < 8; ++j) { a.h[j] = r1[j]; a.h[j + 8] = r2[j]; }
#pragma unroll
        for (int vt = 0; vt < 4; ++vt)
            acc2[vt] = __builtin_amdgcn_wmma_f32_16x16x32_f16(
                false, a.v, false, W1[(kI * 4 + vt) * 32 + tid], (short)0, acc2[vt], false, false);
    }
#pragma unroll
    for (int vt = 0; vt < 4; ++vt) {
        union { v8f v; float f[8]; } c; c.v = acc2[vt];
#pragma unroll
        for (int r = 0; r < 8; ++r) wbuf[r + 8 * half][vt * 16 + m] = c.f[r];
    }
    __syncthreads();
}

// ---------------------------------------------------------------------------
// Edge stage 1: msg_s = w0*f[src]; msg_v = (w1*f[src])⊗Y1 ; scatter-add
// ---------------------------------------------------------------------------
__global__ void edge_stage1(const int* __restrict__ esrc, const int* __restrict__ edst,
                            const float* __restrict__ evec, const float* __restrict__ elen,
                            const _Float16* __restrict__ r1w0p, const _Float16* __restrict__ r1w1p,
                            const float* __restrict__ f,
                            float* __restrict__ agg_s, float* __restrict__ agg_v) {
    __shared__ _Float16 hbuf[16][64];
    __shared__ float wbuf[16][64];
    __shared__ float y1s[16][3];
    int tid = threadIdx.x, tile = blockIdx.x;
    radial_and_y1(tile, tid, elen, evec, r1w0p, r1w1p, hbuf, wbuf, y1s);

    int em = tid >> 1;
    int e2 = tile * 16 + em;
    int s = esrc[e2], d = edst[e2];
    float y0 = y1s[em][0], yy1 = y1s[em][1], y2 = y1s[em][2];
    int u0 = (tid & 1) * 16;

    const float4* fp  = reinterpret_cast<const float4*>(f + (long)s * 32 + u0);
    const float4* wp0 = reinterpret_cast<const float4*>(&wbuf[em][u0]);
    const float4* wp1 = reinterpret_cast<const float4*>(&wbuf[em][32 + u0]);
    float fl[16], w0l[16], w1l[16];
#pragma unroll
    for (int q = 0; q < 4; ++q) {
        float4 fv4 = fp[q], wa4 = wp0[q], wb4 = wp1[q];
        fl[4*q+0]=fv4.x; fl[4*q+1]=fv4.y; fl[4*q+2]=fv4.z; fl[4*q+3]=fv4.w;
        w0l[4*q+0]=wa4.x; w0l[4*q+1]=wa4.y; w0l[4*q+2]=wa4.z; w0l[4*q+3]=wa4.w;
        w1l[4*q+0]=wb4.x; w1l[4*q+1]=wb4.y; w1l[4*q+2]=wb4.z; w1l[4*q+3]=wb4.w;
    }
#pragma unroll
    for (int uu = 0; uu < 16; ++uu) {
        int u = u0 + uu;
        float fsv = fl[uu];
        atomicAdd(&agg_s[(long)d * 32 + u], w0l[uu] * fsv);
        float t = w1l[uu] * fsv;
        long vb = ((long)d * 32 + u) * 3;
        atomicAdd(&agg_v[vb + 0], t * y0);
        atomicAdd(&agg_v[vb + 1], t * yy1);
        atomicAdd(&agg_v[vb + 2], t * y2);
    }
}

// ---------------------------------------------------------------------------
// Edge stage 2: msg_a = wa*fs[src]; msg_b = wb*<fv[src,u,:],Y1>/sqrt(3)
// ---------------------------------------------------------------------------
__global__ void edge_stage2(const int* __restrict__ esrc, const int* __restrict__ edst,
                            const float* __restrict__ evec, const float* __restrict__ elen,
                            const _Float16* __restrict__ r2w0p, const _Float16* __restrict__ r2w1p,
                            const float* __restrict__ fs, const float* __restrict__ fv,
                            float* __restrict__ mid) {
    __shared__ _Float16 hbuf[16][64];
    __shared__ float wbuf[16][64];
    __shared__ float y1s[16][3];
    int tid = threadIdx.x, tile = blockIdx.x;
    radial_and_y1(tile, tid, elen, evec, r2w0p, r2w1p, hbuf, wbuf, y1s);

    int em = tid >> 1;
    int e2 = tile * 16 + em;
    int s = esrc[e2], d = edst[e2];
    float y0 = y1s[em][0], yy1 = y1s[em][1], y2 = y1s[em][2];
    int u0 = (tid & 1) * 16;

    const float4* fp  = reinterpret_cast<const float4*>(fs + (long)s * 32 + u0);
    const float4* fvp = reinterpret_cast<const float4*>(fv + (long)s * 96 + u0 * 3);
    const float4* wp0 = reinterpret_cast<const float4*>(&wbuf[em][u0]);
    const float4* wp1 = reinterpret_cast<const float4*>(&wbuf[em][32 + u0]);
    float fl[16], wal[16], wbl[16], fvl[48];
#pragma unroll
    for (int q = 0; q < 4; ++q) {
        float4 fv4 = fp[q], wa4 = wp0[q], wb4 = wp1[q];
        fl[4*q+0]=fv4.x; fl[4*q+1]=fv4.y; fl[4*q+2]=fv4.z; fl[4*q+3]=fv4.w;
        wal[4*q+0]=wa4.x; wal[4*q+1]=wa4.y; wal[4*q+2]=wa4.z; wal[4*q+3]=wa4.w;
        wbl[4*q+0]=wb4.x; wbl[4*q+1]=wb4.y; wbl[4*q+2]=wb4.z; wbl[4*q+3]=wb4.w;
    }
#pragma unroll
    for (int q = 0; q < 12; ++q) {
        float4 t = fvp[q];
        fvl[4*q+0]=t.x; fvl[4*q+1]=t.y; fvl[4*q+2]=t.z; fvl[4*q+3]=t.w;
    }
#pragma unroll
    for (int uu = 0; uu < 16; ++uu) {
        int u = u0 + uu;
        float dotv = fvl[uu*3+0]*y0 + fvl[uu*3+1]*yy1 + fvl[uu*3+2]*y2;
        atomicAdd(&mid[(long)d * 64 + u], wal[uu] * fl[uu]);
        atomicAdd(&mid[(long)d * 64 + 32 + u], wbl[uu] * dotv * 0.5773502691896258f);
    }
}

// ---------------------------------------------------------------------------
// Host orchestration
// ---------------------------------------------------------------------------
extern "C" void kernel_launch(void* const* d_in, const int* in_sizes, int n_in,
                              void* d_out, int out_size, void* d_ws, size_t ws_size,
                              hipStream_t stream) {
    const int N = 50000, E = 800000;
    const int NT = N / 16;
    const int ET = E / 16;
    const float RN32 = 1.0f / 32.0f;                 // 1/sqrt(32*32)
    const float RN64 = 0.022097086912079608f;        // 1/sqrt(64*32)

    const float* x        = (const float*)d_in[0];
    const float* z        = (const float*)d_in[1];
    const float* node_deg = (const float*)d_in[2];
    const float* evec     = (const float*)d_in[3];
    const float* elen     = (const float*)d_in[4];
    const int*   esrc     = (const int*)  d_in[5];
    const int*   edst     = (const int*)  d_in[6];
    const float* emx_w    = (const float*)d_in[7];
    const float* emx_b    = (const float*)d_in[8];
    const float* emz_w    = (const float*)d_in[9];
    const float* emz_b    = (const float*)d_in[10];
    const float* W_li1    = (const float*)d_in[11];
    const float* W_lm1_s  = (const float*)d_in[12];
    const float* W_lm1_g  = (const float*)d_in[13];
    const float* r1_w0    = (const float*)d_in[14];
    const float* r1_w1    = (const float*)d_in[15];
    const float* W_lo1_s  = (const float*)d_in[16];
    const float* W_lo1_g  = (const float*)d_in[17];
    const float* W_lo1_v  = (const float*)d_in[18];
    const float* W_li2_s  = (const float*)d_in[19];
    const float* W_li2_v  = (const float*)d_in[20];
    const float* W_lm2    = (const float*)d_in[21];
    const float* r2_w0    = (const float*)d_in[22];
    const float* r2_w1    = (const float*)d_in[23];
    const float* W_lo2    = (const float*)d_in[24];
    float* out = (float*)d_out;

    char* base = (char*)d_ws;
    size_t off = 0;
    auto alloc = [&](size_t bytes) -> void* {
        void* p = base + off;
        off += (bytes + 255) & ~(size_t)255;
        return p;
    };
    _Float16* emxp    = (_Float16*)alloc(128  * 32 * 2);
    _Float16* emzp    = (_Float16*)alloc(128  * 32 * 2);
    _Float16* li1p    = (_Float16*)alloc(1024 * 32 * 2);
    _Float16* lm1sp   = (_Float16*)alloc(1024 * 32 * 2);
    _Float16* lm1gp   = (_Float16*)alloc(1024 * 32 * 2);
    _Float16* r1w0p   = (_Float16*)alloc(32   * 64 * 2);
    _Float16* r1w1p   = (_Float16*)alloc(64   * 64 * 2);
    _Float16* lo1sp   = (_Float16*)alloc(1024 * 32 * 2);
    _Float16* lo1gp   = (_Float16*)alloc(1024 * 32 * 2);
    _Float16* lo1vp   = (_Float16*)alloc(1024 * 32 * 2);
    _Float16* li2sp   = (_Float16*)alloc(1024 * 32 * 2);
    _Float16* li2vp   = (_Float16*)alloc(1024 * 32 * 2);
    _Float16* lm2p_lo = (_Float16*)alloc(1024 * 32 * 2);
    _Float16* lm2p_hi = (_Float16*)alloc(1024 * 32 * 2);
    _Float16* r2w0p   = (_Float16*)alloc(32   * 64 * 2);
    _Float16* r2w1p   = (_Float16*)alloc(64   * 64 * 2);
    _Float16* lo2p    = (_Float16*)alloc(2048 * 64 * 2);

    float* inv    = (float*)alloc((size_t)N * 4);
    float* xs     = (float*)alloc((size_t)N * 32 * 4);
    float* zz     = (float*)alloc((size_t)N * 32 * 4);
    float* fbuf   = (float*)alloc((size_t)N * 32 * 4);
    float* mask_s = (float*)alloc((size_t)N * 32 * 4);
    float* mask_g = (float*)alloc((size_t)N * 32 * 4);
    float* agg_s  = (float*)alloc((size_t)N * 32 * 4);
    float* agg_v  = (float*)alloc((size_t)N * 96 * 4);
    float* v1     = (float*)alloc((size_t)N * 96 * 4);
    float* mask2  = (float*)alloc((size_t)N * 64 * 4);
    float* mid    = (float*)alloc((size_t)N * 64 * 4);
    // buffer reuse (lifetimes disjoint):
    float* s1  = xs;      // xs dead after stage-1 fctps
    float* g1  = fbuf;    // f consumed by edge_stage1
    float* fsb = mask_s;  // mask_s consumed by s1 fctp
    float* fvb = agg_v;   // agg_v consumed by v1 fctps

    auto pack = [&](const float* src, _Float16* dst, int K, int V, int srcStride, int Kpad) {
        int T = (Kpad / 32) * (V / 16) * 32;
        pack_weight<<<(T + 255) / 256, 256, 0, stream>>>(src, dst, K, V, srcStride, Kpad);
    };
    auto zero = [&](float* p, int n) {
        zero_kernel<<<(n + 255) / 256, 256, 0, stream>>>(p, n);
    };

    // ---- 0) pack weights to f16 WMMA fragments ----
    pack(emx_w,     emxp,    118,  32, 32, 128);
    pack(emz_w,     emzp,    118,  32, 32, 128);
    pack(W_li1,     li1p,    1024, 32, 32, 1024);
    pack(W_lm1_s,   lm1sp,   1024, 32, 32, 1024);
    pack(W_lm1_g,   lm1gp,   1024, 32, 32, 1024);
    pack(r1_w0,     r1w0p,   10,   64, 64, 32);
    pack(r1_w1,     r1w1p,   64,   64, 64, 64);
    pack(W_lo1_s,   lo1sp,   1024, 32, 32, 1024);
    pack(W_lo1_g,   lo1gp,   1024, 32, 32, 1024);
    pack(W_lo1_v,   lo1vp,   1024, 32, 32, 1024);
    pack(W_li2_s,   li2sp,   1024, 32, 32, 1024);
    pack(W_li2_v,   li2vp,   1024, 32, 32, 1024);
    pack(W_lm2,     lm2p_lo, 1024, 32, 64, 1024);   // cols 0..31 of [1024,64]
    pack(W_lm2 + 32,lm2p_hi, 1024, 32, 64, 1024);   // cols 32..63
    pack(r2_w0,     r2w0p,   10,   64, 64, 32);
    pack(r2_w1,     r2w1p,   64,   64, 64, 64);
    pack(W_lo2,     lo2p,    2048, 64, 64, 2048);

    inv_sqrt_kernel<<<(N + 255) / 256, 256, 0, stream>>>(node_deg, inv, N);

    // ---- 1) node embeddings ----
    embed_kernel<<<NT, 32, 0, stream>>>(x, emx_b, emxp, xs);
    embed_kernel<<<NT, 32, 0, stream>>>(z, emz_b, emzp, zz);

    // ---- 2) stage-1 fctps fused: f (mode1), mask_s, mask_g (mode0) ----
    fctp_multi_kernel<3><<<NT, 32, 0, stream>>>(
        xs, zz, li1p, lm1sp, lm1gp, fbuf, mask_s, mask_g,
        32, 32, 32, 1, 0, 0, inv, nullptr, nullptr, RN32);

    // ---- 3) edge pass 1 ----
    zero(agg_s, N * 32);
    zero(agg_v, N * 96);
    edge_stage1<<<ET, 32, 0, stream>>>(esrc, edst, evec, elen, r1w0p, r1w1p, fbuf, agg_s, agg_v);
    scale_rows_kernel<32><<<(N * 32 + 255) / 256, 256, 0, stream>>>(agg_s, N * 32, inv);
    scale_rows_kernel<96><<<(N * 96 + 255) / 256, 256, 0, stream>>>(agg_v, N * 96, inv);

    // ---- 4) stage-1 outputs fused: s1 = silu comb (mode2), g1 = sigm comb (mode3) ----
    fctp_multi_kernel<2><<<NT, 32, 0, stream>>>(
        agg_s, zz, lo1sp, lo1gp, nullptr, s1, g1, nullptr,
        32, 32, 0, 2, 3, 0, nullptr, mask_s, mask_g, RN32);
    fctp_vec3_kernel<<<NT, 32, 0, stream>>>(agg_v, zz, lo1vp, v1, nullptr, RN32, 0);
    gate_v_kernel<<<(N * 32 + 255) / 256, 256, 0, stream>>>(v1, g1, N * 32);

    // ---- 5) stage-2 fctps fused: fs (mode1) + mask2 halves (mode0) ----
    fctp_multi_kernel<3><<<NT, 32, 0, stream>>>(
        s1, zz, li2sp, lm2p_lo, lm2p_hi, fsb, mask2, mask2 + 32,
        32, 64, 64, 1, 0, 0, inv, nullptr, nullptr, RN32);
    fctp_vec3_kernel<<<NT, 32, 0, stream>>>(v1, zz, li2vp, fvb, inv, RN32, 1);

    // ---- 6) edge pass 2 ----
    zero(mid, N * 64);
    edge_stage2<<<ET, 32, 0, stream>>>(esrc, edst, evec, elen, r2w0p, r2w1p, fsb, fvb, mid);
    scale_rows_kernel<64><<<(N * 64 + 255) / 256, 256, 0, stream>>>(mid, N * 64, inv);

    // ---- 7) final: out = C_S*mask2 + C_X*fctp(mid, zz, W_lo2) ----
    fctp_kernel<4><<<NT, 32, 0, stream>>>(mid, 64, 64, zz, lo2p, out, 64, mask2, 64, RN64, 4);
    (void)in_sizes; (void)n_in; (void)out_size; (void)ws_size;
}
